// CNN_Std_29240137351523
// MI455X (gfx1250) — matrix operand order repaired and checked by reading, not compiled, not run on hardware
//
#include <hip/hip_runtime.h>
#include <math.h>

typedef __attribute__((ext_vector_type(16))) _Float16 v16h;
typedef __attribute__((ext_vector_type(8)))  float    v8f;

namespace {
constexpr int LLEN   = 2048;           // sequence length
constexpr int KW     = 100;            // filter taps
constexpr int GLEN   = 3 * KW - 2;     // 298 = len(flip(w1)*flip(w1)*flip(wl))
constexpr int OUTOFF = 1172;           // output window start in x2
constexpr int NOUT   = 2048;

constexpr int FRONT  = 256;            // zero pad in front of LDS x arrays
constexpr int XLDS   = FRONT + LLEN + 320;  // 2624 elements
constexpr int ULEN   = 2592;           // u scratch (f16) in LDS
constexpr int UBASE  = 860;            // u[n] stored at index n-UBASE (keeps b128 alignment)
constexpr int GP     = 320;            // grev[t] = g[GP - t]
constexpr int GRLEN  = 352;
constexpr int N1LO   = 864;            // first stage-1 tile base (covers u[864..3423])
constexpr int NT1    = 10;             // stage-1 tiles of 256 outputs
constexpr int K2LO   = -304;           // stage-2 K window [-304, 15] (g taps 0..297 padded)
}

union __align__(16) Op16 {
  v16h v;
  unsigned int u[8];
  uint4 q[2];
};

union __align__(16) Pack8h {
  _Float16 h[8];
  uint2 d[2];
};

union __align__(32) Pack8f {
  float f[8];
  float4 q[2];
};

static __device__ __forceinline__ v8f wmma16(const Op16& a, const Op16& b, v8f c) {
  // D = A(16x32 f16) x B(32x16 f16) + C(16x16 f32)
  return __builtin_amdgcn_wmma_f32_16x16x32_f16(false, a.v, false, b.v, (short)0, c,
                                                false, false);
}

// A-operand (16-bit, 16x32): lane half h -> 16 contiguous K starting at 16h (v0..7).
// p points (as uint*) at element (base + 16h) of a 4-byte aligned Toeplitz window.
static __device__ __forceinline__ void loadA(Op16& d, const unsigned int* p) {
#pragma unroll
  for (int j = 0; j < 8; ++j) { d.u[j] = p[j]; }
}
// B-operand (16-bit, 32x16): lane half h -> 16 contiguous K starting at 16h; 16B aligned.
static __device__ __forceinline__ void loadB(Op16& d, const _Float16* p) {
  d.q[0] = ((const uint4*)p)[0];
  d.q[1] = ((const uint4*)p)[1];
}

// ---------------------------------------------------------------------------
// Kernel 0: g = flip(w1) (*) flip(w1) (*) flip(wl)  (298 complex taps) -> d_ws
// ---------------------------------------------------------------------------
__global__ void compute_g(const float* __restrict__ w1r, const float* __restrict__ w1i,
                          const float* __restrict__ wlr, const float* __restrict__ wli,
                          float* __restrict__ g_ws) {
  __shared__ float hr[2 * KW - 1], hi[2 * KW - 1];
  const int tid = threadIdx.x;
  for (int n = tid; n < 2 * KW - 1; n += blockDim.x) {
    float sr = 0.f, si = 0.f;
    const int jlo = (n > KW - 1) ? n - (KW - 1) : 0;
    const int jhi = (n < KW - 1) ? n : KW - 1;
    for (int j = jlo; j <= jhi; ++j) {
      const float ar = w1r[KW - 1 - j],       ai = w1i[KW - 1 - j];
      const float br = w1r[KW - 1 - (n - j)], bi = w1i[KW - 1 - (n - j)];
      sr += ar * br - ai * bi;
      si += ar * bi + ai * br;
    }
    hr[n] = sr; hi[n] = si;
  }
  __syncthreads();
  for (int n = tid; n < GLEN; n += blockDim.x) {
    float sr = 0.f, si = 0.f;
    const int tlo = (n > KW - 1) ? n - (KW - 1) : 0;
    const int thi = (n < 2 * KW - 2) ? n : 2 * KW - 2;
    for (int t = tlo; t <= thi; ++t) {
      const float ar = hr[t], ai = hi[t];
      const float br = wlr[KW - 1 - (n - t)], bi = wli[KW - 1 - (n - t)];
      sr += ar * br - ai * bi;
      si += ar * bi + ai * br;
    }
    g_ws[2 * n]     = sr;
    g_ws[2 * n + 1] = si;
  }
}

// ---------------------------------------------------------------------------
// Main kernel: one workgroup per batch row.
//   Stage 1 (WMMA): u = x (*) x over n in [864, 3423], Toeplitz-GEMM per 256-tile.
//   Stage 2 (WMMA): out = | u (*) g |[1172..3219], Toeplitz(g) x u.
// ---------------------------------------------------------------------------
__global__ __launch_bounds__(256) void cnn_main(const float* __restrict__ xr_g,
                                                const float* __restrict__ xi_g,
                                                const float* __restrict__ g_ws,
                                                float* __restrict__ out) {
  __shared__ __align__(16) _Float16 xf_r[XLDS],  xf_i[XLDS];    // forward x (B operand)
  __shared__ __align__(16) _Float16 xv_r[XLDS],  xv_i[XLDS];    // reversed x (A operand)
  __shared__ __align__(16) _Float16 xv1_r[XLDS], xv1_i[XLDS];   // reversed, shifted by 1
  __shared__ __align__(16) _Float16 u_r[ULEN],   u_i[ULEN];     // stage-1 result (f16)
  __shared__ __align__(16) _Float16 gv_r[GRLEN], gv_i[GRLEN];   // reversed g
  __shared__ __align__(16) _Float16 gv1_r[GRLEN], gv1_i[GRLEN]; // reversed g, shifted by 1

  const int b   = blockIdx.x;
  const int tid = threadIdx.x;
  const float* xr = xr_g + (size_t)b * LLEN;
  const float* xi = xi_g + (size_t)b * LLEN;

  // ---- Stage 0: stage x row (f32 -> f16) into padded fwd/rev LDS views ----
  for (int idx = tid; idx < XLDS; idx += 256) {
    const int j  = idx - FRONT;
    const bool v = (j >= 0) & (j < LLEN);
    xf_r[idx] = (_Float16)(v ? xr[j] : 0.f);
    xf_i[idx] = (_Float16)(v ? xi[j] : 0.f);
    xv_r[idx] = (_Float16)(v ? xr[LLEN - 1 - j] : 0.f);
    xv_i[idx] = (_Float16)(v ? xi[LLEN - 1 - j] : 0.f);
    const int j1  = idx + 1 - FRONT;
    const bool v1 = (j1 >= 0) & (j1 < LLEN);
    xv1_r[idx] = (_Float16)(v1 ? xr[LLEN - 1 - j1] : 0.f);
    xv1_i[idx] = (_Float16)(v1 ? xi[LLEN - 1 - j1] : 0.f);
  }
  for (int idx = tid; idx < GRLEN; idx += 256) {
    const int s  = GP - idx;
    const bool v = (s >= 0) & (s < GLEN);
    gv_r[idx] = (_Float16)(v ? g_ws[2 * s] : 0.f);
    gv_i[idx] = (_Float16)(v ? g_ws[2 * s + 1] : 0.f);
    const int s1  = GP - (idx + 1);
    const bool v1 = (s1 >= 0) & (s1 < GLEN);
    gv1_r[idx] = (_Float16)(v1 ? g_ws[2 * s1] : 0.f);
    gv1_i[idx] = (_Float16)(v1 ? g_ws[2 * s1 + 1] : 0.f);
  }
  __syncthreads();

  // Wave-uniform values pinned to SGPRs so K-loops compile as scalar loops.
  const int wave = __builtin_amdgcn_readfirstlane(tid >> 5);
  const int lane = tid & 31;
  const int hh   = lane >> 4;   // lane half
  const int lm   = lane & 15;   // M for A, N for B/D

  // ---- Stage 1: u[n0+i+16c] = sum_k x[n0+i-k] * x[k+16c] ----
  for (int T = wave; T < NT1; T += 8) {
    const int n0 = N1LO + 256 * T;
    int kstart   = ((n0 - (LLEN - 1)) >> 5) << 5;     // floor to 32
    if (kstart < -FRONT) kstart = -FRONT;
    const int kend    = (n0 + 15 < LLEN - 1) ? (n0 + 15) : (LLEN - 1);
    const int nchunks = __builtin_amdgcn_readfirstlane((kend - kstart + 32) >> 5);

    // A window base into reversed arrays: x[n0+i-k'] = xv[FRONT + 2047 - n0 - i + k']
    const int ibA  = FRONT + (LLEN - 1) - n0 - lm + kstart;
    const int sel  = ibA & 1;  // odd base -> use shifted copy for 4B alignment
    const unsigned int* par =
        (const unsigned int*)((sel ? xv1_r : xv_r) + (ibA - sel) + 16 * hh);
    const unsigned int* pai =
        (const unsigned int*)((sel ? xv1_i : xv_i) + (ibA - sel) + 16 * hh);
    const _Float16* pbr = xf_r + (FRONT + kstart + 16 * lm + 16 * hh);
    const _Float16* pbi = xf_i + (FRONT + kstart + 16 * lm + 16 * hh);

    v8f acc_rr = {}, acc_ii = {}, acc_ri = {};
    for (int it = 0; it < nchunks; ++it) {
      Op16 Ar, Ai, Br, Bi;
      loadA(Ar, par); loadA(Ai, pai);
      loadB(Br, pbr); loadB(Bi, pbi);
      acc_rr = wmma16(Ar, Br, acc_rr);
      acc_ii = wmma16(Ai, Bi, acc_ii);
      acc_ri = wmma16(Ar, Bi, acc_ri);
      acc_ri = wmma16(Ai, Br, acc_ri);   // u_i = xr*xi + xi*xr
      par += 16; pai += 16;              // +32 elements
      pbr += 32; pbi += 32;
    }
    // D: M = r + 8*hh, N = lm -> lane covers u[n0 + 8*hh + 16*lm .. +7] contiguously.
    Pack8h vr, vi;
#pragma unroll
    for (int r = 0; r < 8; ++r) {
      vr.h[r] = (_Float16)(acc_rr[r] - acc_ii[r]);
      vi.h[r] = (_Float16)(acc_ri[r]);
    }
    const int un = n0 + 8 * hh + 16 * lm - UBASE;     // multiple of 4 -> 8B aligned
    ((uint2*)(u_r + un))[0] = vr.d[0];
    ((uint2*)(u_r + un))[1] = vr.d[1];
    ((uint2*)(u_i + un))[0] = vi.d[0];
    ((uint2*)(u_i + un))[1] = vi.d[1];
  }
  __syncthreads();

  // ---- Stage 2: out[m0+i+16c] = | sum_k g[i-k] * u[m0+k+16c] |, K in [-304,15] ----
  {
    const int T  = wave;                 // 8 waves -> 8 tiles
    const int m0 = OUTOFF + 256 * T;

    const int ibA = GP - lm + K2LO;      // grev[GP - i + k'] = g[i - k']
    const int sel = ibA & 1;
    const unsigned int* par =
        (const unsigned int*)((sel ? gv1_r : gv_r) + (ibA - sel) + 16 * hh);
    const unsigned int* pai =
        (const unsigned int*)((sel ? gv1_i : gv_i) + (ibA - sel) + 16 * hh);
    const _Float16* pbr = u_r + (m0 + K2LO + 16 * lm + 16 * hh - UBASE);
    const _Float16* pbi = u_i + (m0 + K2LO + 16 * lm + 16 * hh - UBASE);

    v8f a_rr = {}, a_ii = {}, a_ri = {}, a_ir = {};
#pragma unroll 2
    for (int it = 0; it < 10; ++it) {    // K = 320
      Op16 Ar, Ai, Br, Bi;
      loadA(Ar, par); loadA(Ai, pai);
      loadB(Br, pbr); loadB(Bi, pbi);
      a_rr = wmma16(Ar, Br, a_rr);
      a_ii = wmma16(Ai, Bi, a_ii);
      a_ri = wmma16(Ar, Bi, a_ri);
      a_ir = wmma16(Ai, Br, a_ir);
      par += 16; pai += 16;
      pbr += 32; pbi += 32;
    }
    // Pack the lane's 8 contiguous outputs (32B-aligned) into two b128 stores.
    Pack8f o;
#pragma unroll
    for (int r = 0; r < 8; ++r) {
      const float orr = a_rr[r] - a_ii[r];
      const float oii = a_ri[r] + a_ir[r];
      o.f[r] = sqrtf(orr * orr + oii * oii);
    }
    float* po = out + (size_t)b * NOUT + (256 * T + 8 * hh + 16 * lm);
    ((float4*)po)[0] = o.q[0];
    ((float4*)po)[1] = o.q[1];
  }
}

extern "C" void kernel_launch(void* const* d_in, const int* in_sizes, int n_in,
                              void* d_out, int out_size, void* d_ws, size_t ws_size,
                              hipStream_t stream) {
  (void)in_sizes; (void)n_in; (void)out_size; (void)ws_size;
  const float* xr  = (const float*)d_in[0];
  const float* xi  = (const float*)d_in[1];
  const float* w1r = (const float*)d_in[2];
  const float* w1i = (const float*)d_in[3];
  const float* wlr = (const float*)d_in[4];
  const float* wli = (const float*)d_in[5];
  float* g_ws = (float*)d_ws;            // 298 complex f32 = 2384 bytes

  compute_g<<<1, 128, 0, stream>>>(w1r, w1i, wlr, wli, g_ws);
  cnn_main<<<4096, 256, 0, stream>>>(xr, xi, g_ws, (float*)d_out);
}
